// ProgressiveVQ_33097017983692
// MI455X (gfx1250) — matrix operand order repaired and checked by reading, not compiled, not run on hardware
//
#include <hip/hip_runtime.h>
#include <hip/hip_bf16.h>

#define DATA_DIM   256
#define NUM_CODES  1024
#define NUM_STAGES 8
#define LDS_STRIDE 264   // 256 bf16 + 8 pad -> 132 dwords row stride, conflict-free b128 loads
#define M_TILE     128   // rows per block
#define N_PASS     64    // codes per pass
#define NPASS      (NUM_CODES / N_PASS)   // 16

typedef __attribute__((ext_vector_type(16))) __bf16 v16bf;
typedef __attribute__((ext_vector_type(8)))  float  v8f;

static __device__ __forceinline__ unsigned short f2bf(float f) {
    unsigned u = __float_as_uint(f);
    unsigned r = u + 0x7FFFu + ((u >> 16) & 1u);   // round-to-nearest-even
    return (unsigned short)(r >> 16);
}
static __device__ __forceinline__ unsigned pack2(float a, float b) {
    return (unsigned)f2bf(a) | ((unsigned)f2bf(b) << 16);
}

static __device__ __forceinline__ v8f wmma_bf16(v16bf a, v16bf b, v8f c) {
    return __builtin_amdgcn_wmma_f32_16x16x32_bf16(
        /*neg_a=*/false, a, /*neg_b=*/false, b,
        /*c_mod=*/(short)0, c, /*reuse_a=*/false, /*reuse_b=*/false);
}

union Frag { v16bf v; uint4 q[2]; };

// A fragment (16x32 bf16, M x K): lane<16 holds K {0..7, 16..23}, lane>=16 holds {8..15, 24..31}
static __device__ __forceinline__ v16bf fragA(const unsigned short* base, int kc, int sel8) {
    Frag u;
    const unsigned short* p = base + kc * 32 + sel8;
    u.q[0] = *(const uint4*)(p);
    u.q[1] = *(const uint4*)(p + 16);
    return u.v;
}
// B fragment (32x16 bf16, K x N): lane<16 holds K {0..7, 16..23}, lane>=16 holds {8..15, 24..31}
static __device__ __forceinline__ v16bf fragB(const unsigned short* base, int kc, int sel16) {
    Frag u;
    const unsigned short* p = base + kc * 32 + sel16;
    u.q[0] = *(const uint4*)(p);
    u.q[1] = *(const uint4*)(p + 16);
    return u.v;
}

// CDNA5 async DMA: global -> LDS, 16B per lane, tracked by ASYNCcnt (no VGPR staging).
static __device__ __forceinline__ void async_b128(unsigned lds_off, const unsigned short* gsrc) {
    asm volatile("global_load_async_to_lds_b128 %0, %1, off"
                 :: "v"(lds_off), "v"(gsrc)
                 : "memory");
}

// ---------------------------------------------------------------------------
// Stage prep: bf16-convert one stage's codebook, compute ||c||^2 per code.
// ---------------------------------------------------------------------------
__global__ void __launch_bounds__(256)
vq_prep(const float* __restrict__ cb, unsigned short* __restrict__ cbbf,
        float* __restrict__ cnorm) {
    const int k = blockIdx.x;
    const int d = threadIdx.x;
    float v = cb[(size_t)k * DATA_DIM + d];
    cbbf[(size_t)k * DATA_DIM + d] = f2bf(v);
    float s = v * v;
    #pragma unroll
    for (int m = 16; m > 0; m >>= 1) s += __shfl_xor(s, m, 32);
    __shared__ float ps[8];
    if ((d & 31) == 0) ps[d >> 5] = s;
    __syncthreads();
    if (d == 0) {
        float tot = 0.f;
        #pragma unroll
        for (int i = 0; i < 8; ++i) tot += ps[i];
        cnorm[k] = tot;
    }
}

// ---------------------------------------------------------------------------
// Argmin GEMM: block owns 128 rows; residual = input - recon formed on the fly.
// 8 wave32 = 4 M-pairs x 2 N-pairs; 2x2 wmma_f32_16x16x32_bf16 register blocks.
// B tiles (64 codes/pass) double-buffered via global_load_async_to_lds_b128,
// synchronized with s_wait_asynccnt. dist = ||c||^2 - 2 x.c (||x||^2 is
// row-constant -> irrelevant for argmin).
// ---------------------------------------------------------------------------
__global__ void __launch_bounds__(256)
vq_argmin(const float* __restrict__ input, const float* __restrict__ recon,
          const unsigned short* __restrict__ cbbf, const float* __restrict__ cnorm,
          unsigned int* __restrict__ out_idx, int first) {
    extern __shared__ char smem[];
    unsigned short* lsA  = (unsigned short*)smem;                 // 128 x LDS_STRIDE bf16
    unsigned short* lsB  = lsA + M_TILE * LDS_STRIDE;             // 2 x 64 x LDS_STRIDE bf16
    float*          redD = (float*)(lsB + 2 * N_PASS * LDS_STRIDE);  // 128 x 2
    unsigned int*   redI = (unsigned int*)(redD + M_TILE * 2);       // 128 x 2

    const int  t     = threadIdx.x;
    const int  lane  = t & 31;
    const int  wave  = t >> 5;
    const int  mpair = wave & 3;    // 4 M-pairs of 32 rows
    const int  npair = wave >> 2;   // 2 N-pairs of 32 cols
    const long t0    = (long)blockIdx.x * M_TILE;
    const unsigned lsBoff = (unsigned)(unsigned long long)lsB;    // LDS byte offset

    // ---- kick off async copy of B tile for pass 0 into buffer 0 ----
    #pragma unroll
    for (int i = 0; i < 8; ++i) {
        int ch = t + i * 256;                 // 2048 chunks of 8 bf16
        int k  = ch >> 5;                     // code 0..63
        int c8 = (ch & 31) * 8;
        async_b128(lsBoff + (unsigned)(k * LDS_STRIDE + c8) * 2,
                   cbbf + (size_t)k * DATA_DIM + c8);
    }

    // ---- stage A tile: residual rows -> bf16 LDS (overlaps the async DMA) ----
    #pragma unroll
    for (int i = 0; i < 16; ++i) {
        int ch  = t + i * 256;                // 4096 chunks of 8 floats
        int row = ch >> 5;                    // 0..127
        int c8  = (ch & 31) * 8;
        const float* ip = input + (t0 + row) * DATA_DIM + c8;
        float4 a0 = *(const float4*)(ip);
        float4 a1 = *(const float4*)(ip + 4);
        if (!first) {
            const float* rp = recon + (t0 + row) * DATA_DIM + c8;
            float4 r0 = *(const float4*)(rp);
            float4 r1 = *(const float4*)(rp + 4);
            a0.x -= r0.x; a0.y -= r0.y; a0.z -= r0.z; a0.w -= r0.w;
            a1.x -= r1.x; a1.y -= r1.y; a1.z -= r1.z; a1.w -= r1.w;
        }
        uint4 u;
        u.x = pack2(a0.x, a0.y); u.y = pack2(a0.z, a0.w);
        u.z = pack2(a1.x, a1.y); u.w = pack2(a1.z, a1.w);
        *(uint4*)(lsA + row * LDS_STRIDE + c8) = u;
    }

    float    minD[16];
    unsigned minI[16];
    #pragma unroll
    for (int i = 0; i < 16; ++i) { minD[i] = 3.4e38f; minI[i] = 0u; }

    const unsigned short* pA0 = lsA + (mpair * 32 + (lane & 15)) * LDS_STRIDE;
    const unsigned short* pA1 = pA0 + 16 * LDS_STRIDE;
    const int selA = (lane & 16) ? 8 : 0;
    const int selB = (lane & 16) ? 8 : 0;

    for (int p = 0; p < NPASS; ++p) {
        // ---- issue async copy for pass p+1 into the other buffer ----
        if (p + 1 < NPASS) {
            const int nn = (p + 1) * N_PASS;
            const unsigned dstb = lsBoff + (unsigned)(((p + 1) & 1) * N_PASS * LDS_STRIDE) * 2;
            #pragma unroll
            for (int i = 0; i < 8; ++i) {
                int ch = t + i * 256;
                int k  = ch >> 5;
                int c8 = (ch & 31) * 8;
                async_b128(dstb + (unsigned)(k * LDS_STRIDE + c8) * 2,
                           cbbf + (size_t)(nn + k) * DATA_DIM + c8);
            }
            // 16 outstanding; in-order retire => <=8 means current buffer done
            asm volatile("s_wait_asynccnt 8" ::: "memory");
        } else {
            asm volatile("s_wait_asynccnt 0" ::: "memory");
        }
        __syncthreads();   // current buffer visible to all waves (also covers lsA on p=0)

        const unsigned short* pB0 = lsB + (p & 1) * N_PASS * LDS_STRIDE
                                        + (npair * 32 + (lane & 15)) * LDS_STRIDE;
        const unsigned short* pB1 = pB0 + 16 * LDS_STRIDE;

        v8f a00 = {}, a01 = {}, a10 = {}, a11 = {};
        #pragma unroll
        for (int kc = 0; kc < 8; ++kc) {          // K = 256 in chunks of 32
            v16bf A0 = fragA(pA0, kc, selA);
            v16bf A1 = fragA(pA1, kc, selA);
            v16bf B0 = fragB(pB0, kc, selB);
            v16bf B1 = fragB(pB1, kc, selB);
            a00 = wmma_bf16(A0, B0, a00);
            a01 = wmma_bf16(A0, B1, a01);
            a10 = wmma_bf16(A1, B0, a10);
            a11 = wmma_bf16(A1, B1, a11);
        }

        const int colb = p * N_PASS + npair * 32 + (lane & 15);
        const float cn0 = cnorm[colb];
        const float cn1 = cnorm[colb + 16];
        #pragma unroll
        for (int j = 0; j < 8; ++j) {
            float d;
            d = fmaf(-2.f, a00[j], cn0); if (d < minD[j])     { minD[j]     = d; minI[j]     = colb; }
            d = fmaf(-2.f, a01[j], cn1); if (d < minD[j])     { minD[j]     = d; minI[j]     = colb + 16; }
            d = fmaf(-2.f, a10[j], cn0); if (d < minD[8 + j]) { minD[8 + j] = d; minI[8 + j] = colb; }
            d = fmaf(-2.f, a11[j], cn1); if (d < minD[8 + j]) { minD[8 + j] = d; minI[8 + j] = colb + 16; }
        }
        __syncthreads();   // all waves done reading before next async overwrite
    }

    // ---- per-row argmin: 16-lane xor-shuffle min, then cross-wave LDS reduce ----
    #pragma unroll
    for (int i = 0; i < 16; ++i) {
        float    d  = minD[i];
        unsigned ii = minI[i];
        #pragma unroll
        for (int m = 1; m < 16; m <<= 1) {
            float    od = __shfl_xor(d, m, 32);
            unsigned oi = (unsigned)__shfl_xor((int)ii, m, 32);
            if (od < d) { d = od; ii = oi; }
        }
        if ((lane & 15) == 0) {
            int ml  = i >> 3;
            int j   = i & 7;
            int row = mpair * 32 + ml * 16 + ((lane >> 4) << 3) + j;
            redD[row * 2 + npair] = d;
            redI[row * 2 + npair] = ii;
        }
    }
    __syncthreads();
    if (t < M_TILE) {
        float    best = redD[t * 2];
        unsigned bi   = redI[t * 2];
        float    d1   = redD[t * 2 + 1];
        if (d1 < best) { best = d1; bi = redI[t * 2 + 1]; }
        out_idx[t0 + t] = bi;
    }
}

// ---------------------------------------------------------------------------
// Update: gather q (f32), recon += q, per-block squared-error partial sums.
// 4 rows per 256-thread block; one float4 per thread.
// ---------------------------------------------------------------------------
__global__ void __launch_bounds__(256)
vq_update(const float* __restrict__ input, const float* __restrict__ cb,
          const unsigned int* __restrict__ idx, float* __restrict__ recon,
          float* __restrict__ partials, int first) {
    const int  t = threadIdx.x;
    const long r = (long)blockIdx.x * 4 + (t >> 6);
    const int  c = (t & 63) * 4;
    const size_t off = (size_t)r * DATA_DIM + c;
    const unsigned k = idx[r];
    float4 q = *(const float4*)(cb + (size_t)k * DATA_DIM + c);
    float4 rec;
    if (first) {
        rec = q;
    } else {
        rec = *(const float4*)(recon + off);
        rec.x += q.x; rec.y += q.y; rec.z += q.z; rec.w += q.w;
    }
    *(float4*)(recon + off) = rec;
    float4 in4 = *(const float4*)(input + off);
    float ex = rec.x - in4.x, ey = rec.y - in4.y;
    float ez = rec.z - in4.z, ew = rec.w - in4.w;
    float s = ex * ex + ey * ey + ez * ez + ew * ew;
    #pragma unroll
    for (int m = 16; m > 0; m >>= 1) s += __shfl_xor(s, m, 32);
    __shared__ float ps[8];
    if ((t & 31) == 0) ps[t >> 5] = s;
    __syncthreads();
    if (t == 0) {
        float tot = 0.f;
        #pragma unroll
        for (int i = 0; i < 8; ++i) tot += ps[i];
        partials[blockIdx.x] = tot;
    }
}

// Deterministic fixed-order reduction of block partials -> one MSE slot.
__global__ void __launch_bounds__(256)
vq_reduce(const float* __restrict__ partials, int n,
          float* __restrict__ out, float scale) {
    float s = 0.f;
    for (int i = threadIdx.x; i < n; i += 256) s += partials[i];
    #pragma unroll
    for (int m = 16; m > 0; m >>= 1) s += __shfl_xor(s, m, 32);
    __shared__ float ps[8];
    if ((threadIdx.x & 31) == 0) ps[threadIdx.x >> 5] = s;
    __syncthreads();
    if (threadIdx.x == 0) {
        float tot = 0.f;
        #pragma unroll
        for (int i = 0; i < 8; ++i) tot += ps[i];
        *out = tot * scale;
    }
}

extern "C" void kernel_launch(void* const* d_in, const int* in_sizes, int n_in,
                              void* d_out, int out_size, void* d_ws, size_t ws_size,
                              hipStream_t stream) {
    (void)n_in; (void)out_size; (void)ws_size;
    const float* input     = (const float*)d_in[0];
    const float* codebooks = (const float*)d_in[1];
    const int T = in_sizes[0] / DATA_DIM;   // 131072

    float* recon = (float*)d_out;                       // [T, 256]
    float* mses  = recon + (size_t)T * DATA_DIM;        // [8]

    // workspace: bf16 codebook (512KB) | cnorm (4KB) | idx (512KB) | partials (128KB)
    unsigned short* cbbf   = (unsigned short*)d_ws;
    float*          cnorm  = (float*)(cbbf + (size_t)NUM_CODES * DATA_DIM);
    unsigned int*   idxbuf = (unsigned int*)(cnorm + NUM_CODES);
    float*          parts  = (float*)(idxbuf + T);

    const int   nUpd = T / 4;
    const float invN = 1.0f / ((float)T * (float)DATA_DIM);
    const size_t smem = (size_t)(M_TILE * LDS_STRIDE + 2 * N_PASS * LDS_STRIDE) * sizeof(unsigned short)
                      + (size_t)M_TILE * 2 * (sizeof(float) + sizeof(unsigned int));  // 137216 B

    for (int s = 0; s < NUM_STAGES; ++s) {
        const float* cb = codebooks + (size_t)s * NUM_CODES * DATA_DIM;
        vq_prep  <<<NUM_CODES,  256, 0,    stream>>>(cb, cbbf, cnorm);
        vq_argmin<<<T / M_TILE, 256, smem, stream>>>(input, recon, cbbf, cnorm, idxbuf, s == 0);
        vq_update<<<nUpd,       256, 0,    stream>>>(input, cb, idxbuf, recon, parts, s == 0);
        vq_reduce<<<1,          256, 0,    stream>>>(parts, nUpd, mses + s, invN);
    }
}
